// SphericalFilter_23545010717444
// MI455X (gfx1250) — matrix operand order repaired and checked by reading, not compiled, not run on hardware
//
#include <hip/hip_runtime.h>

typedef __attribute__((ext_vector_type(16))) _Float16 v16h;
typedef __attribute__((ext_vector_type(8)))  float    v8f;

#define N_C             64
#define NUM_BASIS       32
#define WAVES_PER_BLOCK 4
#define EDGES_PER_WAVE  16
#define EDGES_PER_BLOCK (WAVES_PER_BLOCK * EDGES_PER_WAVE)

__global__ __launch_bounds__(128, 1) void spherical_filter_wmma(
    const float* __restrict__ edges,    // [E,3]
    const float* __restrict__ coeffs,   // [64,32]
    const float* __restrict__ centers,  // [32]
    float* __restrict__ out,            // [E,64,5]
    int n_edges)
{
    __shared__ float ylds[WAVES_PER_BLOCK][EDGES_PER_WAVE][5];

    const int tid  = threadIdx.x;
    const int wave = tid >> 5;
    const int lane = tid & 31;
    const int g    = lane >> 4;   // half-wave group (selects K range per ISA layout)
    const int mrow = lane & 15;   // A-matrix row (edge) / C-matrix column (channel)

    const int base = (blockIdx.x * WAVES_PER_BLOCK + wave) * EDGES_PER_WAVE;

    // ---- load this lane's edge (row M=mrow of the A tile), compute radius ----
    int e  = base + mrow;
    int ec = (e < n_edges) ? e : (n_edges > 0 ? n_edges - 1 : 0);
    float ex = edges[ec * 3 + 0];
    float ey = edges[ec * 3 + 1];
    float ez = edges[ec * 3 + 2];
    float r2 = ex * ex + ey * ey + ez * ez;
    float r  = sqrtf(r2);

    // ---- l=2 real spherical harmonics; nan_to_num(0/0) -> 0 at r==0 ----
    if (g == 0) {
        const float c1 = 1.0925484305920792f;   // sqrt(15/(4*pi))
        const float c0 = 0.31539156525252005f;  // sqrt(5/(16*pi))
        const float c2 = 0.5462742152960396f;   // sqrt(15/(16*pi))
        float inv = (r2 > 0.0f) ? (1.0f / r2) : 0.0f;
        ylds[wave][mrow][0] = c1 * ex * ey * inv;
        ylds[wave][mrow][1] = c1 * ey * ez * inv;
        ylds[wave][mrow][2] = c0 * (3.0f * ez * ez - r2) * inv;
        ylds[wave][mrow][3] = c1 * ex * ez * inv;
        ylds[wave][mrow][4] = c2 * (ex * ex - ey * ey) * inv;
    }

    // ---- RBF expansion -> A fragment ----
    // spread = (max(centers)-min(centers))/NUM_BASIS ; linspace is increasing.
    float spread = (centers[NUM_BASIS - 1] - centers[0]) / (float)NUM_BASIS;

    // Lane group g holds K = {g*16 .. g*16+15} in halves 0..15 (matches B layout).
    float cen[16];
    {
        const float4* p0 = reinterpret_cast<const float4*>(centers + g * 16);      // 16B aligned
        const float4* p1 = reinterpret_cast<const float4*>(centers + g * 16 + 8);  // 16B aligned
        float4 a0 = p0[0], a1 = p0[1], b0 = p1[0], b1 = p1[1];
        cen[0]  = a0.x; cen[1]  = a0.y; cen[2]  = a0.z; cen[3]  = a0.w;
        cen[4]  = a1.x; cen[5]  = a1.y; cen[6]  = a1.z; cen[7]  = a1.w;
        cen[8]  = b0.x; cen[9]  = b0.y; cen[10] = b0.z; cen[11] = b0.w;
        cen[12] = b1.x; cen[13] = b1.y; cen[14] = b1.z; cen[15] = b1.w;
    }
    v16h afrag;
#pragma unroll
    for (int h = 0; h < 16; ++h) {
        float d = r - cen[h];
        afrag[h] = (_Float16)__expf(-spread * d * d);
    }

    // ---- B fragments: B[k][c] = coeffs[c][k] ----
    // Tile t: lane column c = t*16 + mrow; half h -> K = g*16 + h (contiguous K order, same as A).
    v16h bfrag[4];
#pragma unroll
    for (int t = 0; t < 4; ++t) {
        int c = t * 16 + mrow;
        const float4* p = reinterpret_cast<const float4*>(coeffs + c * NUM_BASIS + g * 16);
#pragma unroll
        for (int q = 0; q < 4; ++q) {
            float4 v = p[q];
            bfrag[t][q * 4 + 0] = (_Float16)v.x;
            bfrag[t][q * 4 + 1] = (_Float16)v.y;
            bfrag[t][q * 4 + 2] = (_Float16)v.z;
            bfrag[t][q * 4 + 3] = (_Float16)v.w;
        }
    }

    // ---- rc = rbfs @ coeffs^T : 4 WMMA tiles (16 edges x 64 channels, K=32) ----
    // EXEC is all-ones here (no divergent returns above).
    v8f acc[4];
#pragma unroll
    for (int t = 0; t < 4; ++t) {
        v8f z = {};
        acc[t] = __builtin_amdgcn_wmma_f32_16x16x32_f16(
            /*neg_a=*/false, afrag, /*neg_b=*/false, bfrag[t],
            /*c_mod=*/(short)0, z, /*reuse_a=*/false, /*reuse_b=*/false);
    }

    __syncthreads();

    // ---- outer product with harmonics + store ----
    // C/D layout: lane holds column N = mrow; VGPR rr holds row M = g*8 + rr.
#pragma unroll
    for (int rr = 0; rr < 8; ++rr) {
        int M  = g * 8 + rr;
        int eo = base + M;
        if (eo < n_edges) {
            float y0 = ylds[wave][M][0];   // broadcast LDS reads (conflict-free)
            float y1 = ylds[wave][M][1];
            float y2 = ylds[wave][M][2];
            float y3 = ylds[wave][M][3];
            float y4 = ylds[wave][M][4];
            size_t obase = (size_t)eo * (N_C * 5);
#pragma unroll
            for (int t = 0; t < 4; ++t) {
                int   c  = t * 16 + mrow;
                float rc = acc[t][rr];
                float* o = out + obase + (size_t)c * 5;
                o[0] = rc * y0;
                o[1] = rc * y1;
                o[2] = rc * y2;
                o[3] = rc * y3;
                o[4] = rc * y4;
            }
        }
    }
}

extern "C" void kernel_launch(void* const* d_in, const int* in_sizes, int n_in,
                              void* d_out, int out_size, void* d_ws, size_t ws_size,
                              hipStream_t stream) {
    (void)n_in; (void)out_size; (void)d_ws; (void)ws_size;
    const float* edges   = (const float*)d_in[0];
    const float* coeffs  = (const float*)d_in[1];
    const float* centers = (const float*)d_in[2];
    float* out = (float*)d_out;

    int n_edges = in_sizes[0] / 3;
    int blocks  = (n_edges + EDGES_PER_BLOCK - 1) / EDGES_PER_BLOCK;
    if (blocks < 1) blocks = 1;
    spherical_filter_wmma<<<blocks, 128, 0, stream>>>(edges, coeffs, centers, out, n_edges);
}
